// bid_mpgnn_64793876627816
// MI455X (gfx1250) — matrix operand order repaired and checked by reading, not compiled, not run on hardware
//
#include <hip/hip_runtime.h>
#include <hip/hip_bf16.h>
#include <math.h>

// ---------------------------------------------------------------------------
// Problem constants (match reference: L=8, M=32768, K=8, H=128, F=32, D=3)
// ---------------------------------------------------------------------------
#define LFR   8
#define MNODE 32768
#define KDEG  8
#define HDIM  128
#define FEAT  32

typedef __attribute__((ext_vector_type(16))) __bf16 v16bf;
typedef __attribute__((ext_vector_type(8)))  __bf16 v8bf;
typedef __attribute__((ext_vector_type(8)))  float  v8f;

union V16U { v16bf v; v8bf h[2]; };

// ---------------------------------------------------------------------------
// Hardware tanh (gfx1250 TRANS32 op). A trailing v_nop (or any independent
// op) satisfies the 1-op TRANS hazard the compiler can't see through asm.
// ---------------------------------------------------------------------------
__device__ inline float htanh(float x) {
    float r;
    asm("v_tanh_f32 %0, %1\n\tv_nop" : "=v"(r) : "v"(x));
    return r;
}

// Batched tanh over 8 values: consecutive independent v_tanh ops cover each
// other's TRANS hazard; only one trailing v_nop is needed for the last one.
__device__ inline v8f htanh8(v8f a) {
    float a0 = a[0], a1 = a[1], a2 = a[2], a3 = a[3];
    float a4 = a[4], a5 = a[5], a6 = a[6], a7 = a[7];
    asm("v_tanh_f32 %0, %0\n\t"
        "v_tanh_f32 %1, %1\n\t"
        "v_tanh_f32 %2, %2\n\t"
        "v_tanh_f32 %3, %3\n\t"
        "v_tanh_f32 %4, %4\n\t"
        "v_tanh_f32 %5, %5\n\t"
        "v_tanh_f32 %6, %6\n\t"
        "v_tanh_f32 %7, %7\n\t"
        "v_nop"
        : "+v"(a0), "+v"(a1), "+v"(a2), "+v"(a3),
          "+v"(a4), "+v"(a5), "+v"(a6), "+v"(a7));
    v8f r;
    r[0] = a0; r[1] = a1; r[2] = a2; r[3] = a3;
    r[4] = a4; r[5] = a5; r[6] = a6; r[7] = a7;
    return r;
}

// ---------------------------------------------------------------------------
// One 16x16 output tile: acc = bias; for i<NSTEPS: acc += A[i] x B[i].
// B fragments are software-pipelined one K-chunk ahead of the WMMA.
// wrow already points at  wt + (nb+ln)*KIN + lh*8 ; chunk i lives at +i*32.
// ---------------------------------------------------------------------------
template<int NSTEPS>
__device__ inline v8f gemm_acc(const V16U* a, const __bf16* __restrict__ wrow, float bv)
{
    v8f acc;
    #pragma unroll
    for (int v = 0; v < 8; ++v) acc[v] = bv;
    V16U b0, b1;
    b0.h[0] = *(const v8bf*)(wrow);
    b0.h[1] = *(const v8bf*)(wrow + 16);
    #pragma unroll
    for (int i = 0; i < NSTEPS; ++i) {
        if (i + 1 < NSTEPS) {
            const __bf16* wn = wrow + (i + 1) * 32;
            b1.h[0] = *(const v8bf*)(wn);
            b1.h[1] = *(const v8bf*)(wn + 16);
        }
        acc = __builtin_amdgcn_wmma_f32_16x16x32_bf16(false, a[i].v, false, b0.v,
                                                      (short)0, acc, false, false);
        b0 = b1;
    }
    return acc;
}

// ---------------------------------------------------------------------------
// Fused resnet block:  out = tanh( (tanh(h1@w2+b2) + x) @ w3 + b3 )
//                      h1  = tanh( x@w1 + b1 )
// x = concat(xlo, xhi)  (xhi==null -> zeros; KIN==128 -> x = xlo)
// Weights are pre-transposed bf16: wNt is [Nout][Kin] row-major.
// Each wave owns 16 rows and runs all 3 stages privately (row-local math):
// no inter-wave barriers. WMMA: v_wmma_f32_16x16x32_bf16.
// ---------------------------------------------------------------------------
template<int KIN>
__global__ __launch_bounds__(128)
void resnet_wmma_kernel(const __bf16* __restrict__ xlo,
                        const __bf16* __restrict__ xhi,
                        const __bf16* __restrict__ w1t, const float* __restrict__ b1,
                        const __bf16* __restrict__ w2t, const float* __restrict__ b2,
                        const __bf16* __restrict__ w3t, const float* __restrict__ b3,
                        __bf16* __restrict__ out_bf,
                        float*  __restrict__ out_f32)
{
    constexpr int WAVES  = 4;
    constexpr int KSTEPS = KIN / 32;
    constexpr int HSTEPS = (KIN == 256) ? KSTEPS / 2 : KSTEPS;  // lower-half-only count
    __shared__ __align__(16) __bf16 lds[WAVES * 2 * 16 * KIN];   // <= 64KB

    const int wave = threadIdx.x >> 5;
    const int lane = threadIdx.x & 31;
    const int lh   = lane >> 4;      // half-wave select
    const int ln   = lane & 15;

    __bf16* H1 = lds + wave * (2 * 16 * KIN);
    __bf16* H2 = H1 + 16 * KIN;

    const int row0 = blockIdx.x * (WAVES * 16) + wave * 16;
    const bool lowhalf_only = (KIN == 256) && (xhi == nullptr);

    // ---------------- stage 1: H1 = tanh(x @ w1 + b1) ----------------
    {
        V16U a[KSTEPS];
        #pragma unroll
        for (int i = 0; i < KSTEPS; ++i) {
            const int k0 = i * 32;
            if (k0 < 128 || !lowhalf_only) {
                const __bf16* xs = (k0 < 128)
                    ? (xlo + (size_t)(row0 + ln) * HDIM + k0)
                    : (xhi + (size_t)(row0 + ln) * HDIM + (k0 - 128));
                a[i].h[0] = *(const v8bf*)(xs + lh * 8);       // K {0..7}/{8..15}
                a[i].h[1] = *(const v8bf*)(xs + lh * 8 + 16);  // K {16..23}/{24..31}
            }
        }
        for (int nt = 0; nt < KIN / 16; ++nt) {
            const int nb = nt * 16;
            const float bv = b1[nb + ln];
            const __bf16* wrow = w1t + (size_t)(nb + ln) * KIN + lh * 8;
            v8f acc;
            if (lowhalf_only) acc = gemm_acc<HSTEPS>(a, wrow, bv);
            else              acc = gemm_acc<KSTEPS>(a, wrow, bv);
            acc = htanh8(acc);
            #pragma unroll
            for (int v = 0; v < 8; ++v)
                H1[(v + lh * 8) * KIN + nb + ln] = (__bf16)acc[v];
        }
    }

    // ---------------- stage 2: H2 = tanh(H1 @ w2 + b2) + x ----------------
    {
        V16U a[KSTEPS];
        #pragma unroll
        for (int i = 0; i < KSTEPS; ++i) {
            const __bf16* as = H1 + ln * KIN + i * 32 + lh * 8;
            a[i].h[0] = *(const v8bf*)(as);
            a[i].h[1] = *(const v8bf*)(as + 16);
        }
        for (int nt = 0; nt < KIN / 16; ++nt) {
            const int nb = nt * 16;
            const __bf16* wrow = w2t + (size_t)(nb + ln) * KIN + lh * 8;
            v8f acc = gemm_acc<KSTEPS>(a, wrow, b2[nb + ln]);
            acc = htanh8(acc);
            #pragma unroll
            for (int v = 0; v < 8; ++v) {
                const int r = v + lh * 8;
                const int n = nb + ln;
                float xv;
                if constexpr (KIN == 128) {
                    xv = (float)xlo[(size_t)(row0 + r) * HDIM + n];
                } else {
                    if (n < 128)   xv = (float)xlo[(size_t)(row0 + r) * HDIM + n];
                    else if (xhi)  xv = (float)xhi[(size_t)(row0 + r) * HDIM + (n - 128)];
                    else           xv = 0.0f;
                }
                H2[r * KIN + n] = (__bf16)(acc[v] + xv);
            }
        }
    }

    // ---------------- stage 3: out = tanh(H2 @ w3 + b3) ----------------
    {
        V16U a[KSTEPS];
        #pragma unroll
        for (int i = 0; i < KSTEPS; ++i) {
            const __bf16* as = H2 + ln * KIN + i * 32 + lh * 8;
            a[i].h[0] = *(const v8bf*)(as);
            a[i].h[1] = *(const v8bf*)(as + 16);
        }
        for (int nt = 0; nt < HDIM / 16; ++nt) {
            const int nb = nt * 16;
            const __bf16* wrow = w3t + (size_t)(nb + ln) * KIN + lh * 8;
            v8f acc = gemm_acc<KSTEPS>(a, wrow, b3[nb + ln]);
            acc = htanh8(acc);
            if (out_f32) {
                #pragma unroll
                for (int v = 0; v < 8; ++v)
                    out_f32[(size_t)(row0 + v + lh * 8) * HDIM + nb + ln] = acc[v];
            } else {
                #pragma unroll
                for (int v = 0; v < 8; ++v)
                    out_bf[(size_t)(row0 + v + lh * 8) * HDIM + nb + ln] = (__bf16)acc[v];
            }
        }
    }
}

// ---------------------------------------------------------------------------
// Gather K=8 predecessor rows from prev front (f32) and sum -> bf16 [M][128]
// ---------------------------------------------------------------------------
__global__ __launch_bounds__(256)
void gather_sum_kernel(const float* __restrict__ prev,
                       const int*   __restrict__ idx,
                       __bf16*      __restrict__ sbf)
{
    const int t = blockIdx.x * blockDim.x + threadIdx.x;   // M*32 threads
    const int m = t >> 5;
    const int c = (t & 31) << 2;
    float4 s = make_float4(0.f, 0.f, 0.f, 0.f);
    #pragma unroll
    for (int k = 0; k < KDEG; ++k) {
        const int p = idx[m * KDEG + k];
        const float4 f = *(const float4*)(prev + (size_t)p * HDIM + c);
        s.x += f.x; s.y += f.y; s.z += f.z; s.w += f.w;
    }
    __bf16* o = sbf + (size_t)m * HDIM + c;
    o[0] = (__bf16)s.x; o[1] = (__bf16)s.y; o[2] = (__bf16)s.z; o[3] = (__bf16)s.w;
}

// ---------------------------------------------------------------------------
// Embed: tanh(feats[m,:] @ W[32,128] + b)  (tiny, VALU + v_tanh)
// ---------------------------------------------------------------------------
__global__ __launch_bounds__(256)
void embed_kernel(const float* __restrict__ feats,  // [M][32]
                  const float* __restrict__ w,      // [32][128]
                  const float* __restrict__ b,      // [128]
                  __bf16* __restrict__ out_bf,
                  float*  __restrict__ out_f32)
{
    const int t = blockIdx.x * blockDim.x + threadIdx.x;   // M*128 threads
    const int m = t >> 7;
    const int n = t & 127;
    float acc = b[n];
    #pragma unroll
    for (int k = 0; k < FEAT; ++k)
        acc = fmaf(feats[(size_t)m * FEAT + k], w[k * HDIM + n], acc);
    const float val = htanh(acc);
    const size_t o = (size_t)m * HDIM + n;
    if (out_bf)  out_bf[o]  = (__bf16)val;
    if (out_f32) out_f32[o] = val;
}

// ---------------------------------------------------------------------------
// Weight prep: batched transpose + f32->bf16.  w: [B][K][N] -> wt: [B][N][K]
// ---------------------------------------------------------------------------
__global__ __launch_bounds__(256)
void wprep_kernel(const float* __restrict__ w, __bf16* __restrict__ wt,
                  int K, int N, long total)
{
    const long t = (long)blockIdx.x * blockDim.x + threadIdx.x;
    if (t >= total) return;
    const long kn = (long)K * N;
    const long bb = t / kn;
    const long r  = t - bb * kn;
    const int k = (int)(r / N);
    const int n = (int)(r - (long)k * N);
    wt[bb * kn + (long)n * K + k] = (__bf16)w[t];
}

// ---------------------------------------------------------------------------
extern "C" void kernel_launch(void* const* d_in, const int* in_sizes, int n_in,
                              void* d_out, int out_size, void* d_ws, size_t ws_size,
                              hipStream_t stream)
{
    (void)in_sizes; (void)n_in; (void)out_size; (void)ws_size;

    const float* node_feats = (const float*)d_in[0];
    const int*   pred_idx   = (const int*)d_in[1];
    const float* embed_w    = (const float*)d_in[2];
    const float* embed_b    = (const float*)d_in[3];
    const float* mp_w1 = (const float*)d_in[4];  const float* mp_b1 = (const float*)d_in[5];
    const float* mp_w2 = (const float*)d_in[6];  const float* mp_b2 = (const float*)d_in[7];
    const float* mp_w3 = (const float*)d_in[8];  const float* mp_b3 = (const float*)d_in[9];
    const float* cw_w1 = (const float*)d_in[10]; const float* cw_b1 = (const float*)d_in[11];
    const float* cw_w2 = (const float*)d_in[12]; const float* cw_b2 = (const float*)d_in[13];
    const float* cw_w3 = (const float*)d_in[14]; const float* cw_b3 = (const float*)d_in[15];
    const float* ch_w1 = (const float*)d_in[16]; const float* ch_b1 = (const float*)d_in[17];
    const float* ch_w2 = (const float*)d_in[18]; const float* ch_b2 = (const float*)d_in[19];
    const float* ch_w3 = (const float*)d_in[20]; const float* ch_b3 = (const float*)d_in[21];
    const float* nw_w1 = (const float*)d_in[22]; const float* nw_b1 = (const float*)d_in[23];
    const float* nw_w2 = (const float*)d_in[24]; const float* nw_b2 = (const float*)d_in[25];
    const float* nw_w3 = (const float*)d_in[26]; const float* nw_b3 = (const float*)d_in[27];
    const float* nh_w1 = (const float*)d_in[28]; const float* nh_b1 = (const float*)d_in[29];
    const float* nh_w2 = (const float*)d_in[30]; const float* nh_b2 = (const float*)d_in[31];
    const float* nh_w3 = (const float*)d_in[32]; const float* nh_b3 = (const float*)d_in[33];

    // ---- carve workspace (256B aligned) ----
    char* wsb = (char*)d_ws;
    size_t off = 0;
    auto alloc = [&](size_t elems) -> __bf16* {
        __bf16* p = (__bf16*)(wsb + off);
        off += ((elems * sizeof(__bf16) + 255) / 256) * 256;
        return p;
    };
    __bf16* mpw1t = alloc(6 * 128 * 128);
    __bf16* mpw2t = alloc(6 * 128 * 128);
    __bf16* mpw3t = alloc(6 * 128 * 128);
    __bf16* cww1t = alloc(3 * 256 * 256);
    __bf16* cww2t = alloc(3 * 256 * 256);
    __bf16* cww3t = alloc(3 * 128 * 256);
    __bf16* chw1t = alloc(3 * 128 * 128);
    __bf16* chw2t = alloc(3 * 128 * 128);
    __bf16* chw3t = alloc(3 * 128 * 128);
    __bf16* nww1t = alloc(3 * 256 * 256);
    __bf16* nww2t = alloc(3 * 256 * 256);
    __bf16* nww3t = alloc(3 * 128 * 256);
    __bf16* nhw1t = alloc(3 * 128 * 128);
    __bf16* nhw2t = alloc(3 * 128 * 128);
    __bf16* nhw3t = alloc(3 * 128 * 128);
    const size_t MH = (size_t)MNODE * HDIM;
    __bf16* bufS  = alloc(MH);
    __bf16* bufT1 = alloc(MH);
    __bf16* bufT2 = alloc(MH);
    __bf16* bufE  = alloc(MH);

    // ---- weight prep (once; tiny) ----
    auto wprep = [&](const float* w, __bf16* wt, int B, int K, int N) {
        const long total = (long)B * K * N;
        wprep_kernel<<<(int)((total + 255) / 256), 256, 0, stream>>>(w, wt, K, N, total);
    };
    wprep(mp_w1, mpw1t, 6, 128, 128);
    wprep(mp_w2, mpw2t, 6, 128, 128);
    wprep(mp_w3, mpw3t, 6, 128, 128);
    wprep(cw_w1, cww1t, 3, 256, 256);
    wprep(cw_w2, cww2t, 3, 256, 256);
    wprep(cw_w3, cww3t, 3, 256, 128);
    wprep(ch_w1, chw1t, 3, 128, 128);
    wprep(ch_w2, chw2t, 3, 128, 128);
    wprep(ch_w3, chw3t, 3, 128, 128);
    wprep(nw_w1, nww1t, 3, 256, 256);
    wprep(nw_w2, nww2t, 3, 256, 256);
    wprep(nw_w3, nww3t, 3, 256, 128);
    wprep(nh_w1, nhw1t, 3, 128, 128);
    wprep(nh_w2, nhw2t, 3, 128, 128);
    wprep(nh_w3, nhw3t, 3, 128, 128);

    float* out = (float*)d_out;

    // ---- front 0: output = tanh(embed(feats[0])) (f32 straight to d_out) ----
    embed_kernel<<<MNODE * HDIM / 256, 256, 0, stream>>>(
        node_feats, embed_w, embed_b, nullptr, out);

    const int rgrid = MNODE / 64;   // 4 waves x 16 rows per block
    auto resnet128 = [&](const __bf16* x,
                         const __bf16* w1, const float* bb1,
                         const __bf16* w2, const float* bb2,
                         const __bf16* w3, const float* bb3,
                         __bf16* ob, float* of) {
        resnet_wmma_kernel<128><<<rgrid, 128, 0, stream>>>(x, nullptr,
            w1, bb1, w2, bb2, w3, bb3, ob, of);
    };
    auto resnet256 = [&](const __bf16* xl, const __bf16* xh,
                         const __bf16* w1, const float* bb1,
                         const __bf16* w2, const float* bb2,
                         const __bf16* w3, const float* bb3,
                         __bf16* ob, float* of) {
        resnet_wmma_kernel<256><<<rgrid, 128, 0, stream>>>(xl, xh,
            w1, bb1, w2, bb2, w3, bb3, ob, of);
    };

    // ---- fronts 1..7 (sequential dependency via gather of prev front) ----
    for (int l = 1; l < LFR; ++l) {
        const int d = (l - 1 < 2) ? (l - 1) : 2;
        const float* prev = out + (size_t)(l - 1) * MH;

        // mailbox gather + sum -> bf16
        gather_sum_kernel<<<MNODE * 32 / 256, 256, 0, stream>>>(
            prev, pred_idx + (size_t)(l - 1) * MNODE * KDEG, bufS);

        // mp resnet pair (128-wide)
        const size_t m0 = (size_t)(d * 2 + 0) * 128 * 128;
        const size_t m1 = (size_t)(d * 2 + 1) * 128 * 128;
        resnet128(bufS,  mpw1t + m0, mp_b1 + (d * 2 + 0) * 128,
                         mpw2t + m0, mp_b2 + (d * 2 + 0) * 128,
                         mpw3t + m0, mp_b3 + (d * 2 + 0) * 128, bufT1, nullptr);
        resnet128(bufT1, mpw1t + m1, mp_b1 + (d * 2 + 1) * 128,
                         mpw2t + m1, mp_b2 + (d * 2 + 1) * 128,
                         mpw3t + m1, mp_b3 + (d * 2 + 1) * 128, bufT2, nullptr);

        // embed this front (bf16 only)
        embed_kernel<<<MNODE * HDIM / 256, 256, 0, stream>>>(
            node_feats + (size_t)l * MNODE * FEAT, embed_w, embed_b, bufE, nullptr);

        // comb branch: x = cat(embed, zeros) -> wide -> h
        const size_t wW = (size_t)d * 256 * 256, w3W = (size_t)d * 128 * 256;
        const size_t hW = (size_t)d * 128 * 128;
        resnet256(bufE, nullptr,
                  cww1t + wW, cw_b1 + d * 256,
                  cww2t + wW, cw_b2 + d * 256,
                  cww3t + w3W, cw_b3 + d * 128, bufT1, nullptr);
        resnet128(bufT1, chw1t + hW, ch_b1 + d * 128,
                         chw2t + hW, ch_b2 + d * 128,
                         chw3t + hW, ch_b3 + d * 128, bufS, nullptr);

        // node branch: x = cat(e, r) -> wide -> h -> f32 output front l
        resnet256(bufS, bufT2,
                  nww1t + wW, nw_b1 + d * 256,
                  nww2t + wW, nw_b2 + d * 256,
                  nww3t + w3W, nw_b3 + d * 128, bufT1, nullptr);
        resnet128(bufT1, nhw1t + hW, nh_b1 + d * 128,
                         nhw2t + hW, nh_b2 + d * 128,
                         nhw3t + hW, nh_b3 + d * 128,
                  nullptr, out + (size_t)l * MH);
    }
}
